// GenSupConLoss_86655260164850
// MI455X (gfx1250) — matrix operand order repaired and checked by reading, not compiled, not run on hardware
//
#include <hip/hip_runtime.h>
#include <hip/hip_bf16.h>
#include <stdint.h>

// ---------------------------------------------------------------------------
// SupCon loss, fused flash-style pass for MI455X (gfx1250, wave32).
//   N = 8192 rows (4096 samples x 2 views), D = 128, C = 100 labels.
// log2-domain logits: features pre-scaled by sqrt(log2(e)/0.07) so that
//   S' = S*log2(e), exp(S-m) == exp2(S'-m')  -> bare v_exp_f32, no mul.
// Row stats (m', Z, P', Msum) kept per lane, merged across 16 column-lanes,
// then across 8 j-slices (split-K for occupancy), all deterministic.
// Matmuls: V_WMMA_F32_16X16X32_BF16 with 2-term bf16 split (hi*hi+hi*lo+lo*hi).
// Diagonal masking is hoisted out of the hot path: only the single 16-aligned
// tile with j0 == i0 can contain diagonal elements (wave-uniform branch).
// ---------------------------------------------------------------------------

#define N_TOT   8192
#define D_FEAT  128
#define C_LAB   100
#define JSPLIT  8
#define JSLICE  (N_TOT / JSPLIT)            // 1024 columns per slice
#define SCALE_LOG2 4.5398165f               // sqrt(log2(e)/0.07)
#define LN2 0.6931471805599453f

typedef __attribute__((ext_vector_type(16))) __bf16 v16bf;
typedef __attribute__((ext_vector_type(8)))  __bf16 v8bf;
typedef __attribute__((ext_vector_type(8)))  float  v8f;

union V16U { v16bf v; v8bf h[2]; };

// B operand (32x16 bf16, K x N): lane l -> column j0+(l&15); K-half = (l>>4)*16.
__device__ __forceinline__ v16bf load_b_tile(const __bf16* __restrict__ base,
                                             int j0, int kb, int nIdx, int kh) {
  const __bf16* p = base + (size_t)(j0 + nIdx) * D_FEAT + kb * 32 + kh * 16;
  return *(const v16bf*)p;
}

// A operand (16x32 bf16, M x K): lane l -> row i0+(l&15); two contiguous 16B runs.
__device__ __forceinline__ v16bf load_a_tile(const __bf16* __restrict__ base,
                                             int i0, int kb, int mIdx, int kh) {
  const __bf16* p = base + (size_t)(i0 + mIdx) * D_FEAT + kb * 32 + kh * 16;
  V16U u;
  u.h[0] = *(const v8bf*)p;
  u.h[1] = *(const v8bf*)(p + 8);
  return u.v;
}

// Online-softmax update for one 16x16 tile (8 rows per lane).
// DIAG=false is the hot path: no compares / cndmasks at all.
template <bool DIAG>
__device__ __forceinline__ void tile_update(const v8f& accS, const v8f& accM,
                                            int i0, int colg, int kh,
                                            float* m, float* Z,
                                            float* P, float* Ms) {
#pragma unroll
  for (int r = 0; r < 8; r++) {
    float S  = accS[r];                     // logit, log2 domain
    float mk = accM[r];
    bool diag = false;
    if (DIAG) {
      const int rowg = i0 + r + 8 * kh;     // lanes 16-31 are rows r+8
      diag = (rowg == colg);
      if (diag) mk = 0.f;                   // logits_mask zeroes diagonal
    }
    P[r]  = fmaf(mk, S, P[r]);
    Ms[r] += mk;
    float mn = fmaxf(m[r], S);              // rowmax includes the diagonal
    float e  = __builtin_amdgcn_exp2f(S - mn);
    if (DIAG && diag) e = 0.f;              // Z excludes the diagonal
    Z[r] = Z[r] * __builtin_amdgcn_exp2f(m[r] - mn) + e;
    m[r] = mn;
  }
}

// ---------------- preprocessing ----------------

__global__ __launch_bounds__(256) void prep_feat(const float* __restrict__ f,
                                                 __bf16* __restrict__ Fh,
                                                 __bf16* __restrict__ Fl,
                                                 int n) {
  int idx = blockIdx.x * blockDim.x + threadIdx.x;
  if (idx < n) {
    float x = f[idx] * SCALE_LOG2;          // logits come out in log2 domain
    __bf16 h = (__bf16)x;
    Fh[idx] = h;
    Fl[idx] = (__bf16)(x - (float)h);
  }
}

__global__ __launch_bounds__(128) void prep_lab(const float* __restrict__ lab,
                                                __bf16* __restrict__ Lb) {
  __shared__ float red[128];
  const int n = blockIdx.x;                 // row; source sample = n>>1
  const int c = threadIdx.x;
  const float* row = lab + (size_t)(n >> 1) * C_LAB;
  float v = (c < C_LAB) ? row[c] : 0.0f;
  red[c] = v * v;
  __syncthreads();
  for (int s = 64; s > 0; s >>= 1) {
    if (c < s) red[c] += red[c + s];
    __syncthreads();
  }
  float inv = rsqrtf(red[0] + 1e-20f);
  Lb[(size_t)n * D_FEAT + c] = (__bf16)(v * inv);
}

// ---------------- fused main pass ----------------
// wave id w in [0,4096): slice q = w>>9, row strip = w&511.
// Waves in a block share the slice and hold 8 consecutive strips (B reuse in WGP$).

__global__ __launch_bounds__(256) void supcon_main(const __bf16* __restrict__ Fh,
                                                   const __bf16* __restrict__ Fl,
                                                   const __bf16* __restrict__ Lb,
                                                   float* __restrict__ stats) {
  const int lane = threadIdx.x & 31;
  const int wave = threadIdx.x >> 5;
  const int wid  = blockIdx.x * 8 + wave;   // 0..4095
  const int q    = wid >> 9;                // j-slice 0..7
  const int strip = wid & 511;              // row strip
  const int i0   = strip * 16;
  const int jBeg = q * JSLICE;
  const int jEnd = jBeg + JSLICE;
  const int mIdx = lane & 15;
  const int kh   = lane >> 4;

  float* Sm = stats;
  float* Sz = Sm + (size_t)JSPLIT * N_TOT;
  float* Sp = Sz + (size_t)JSPLIT * N_TOT;
  float* Ss = Sp + (size_t)JSPLIT * N_TOT;

  // A operands resident in VGPRs for the whole slice loop.
  v16bf aFh[4], aFl[4], aL[4];
#pragma unroll
  for (int kb = 0; kb < 4; kb++) {
    aFh[kb] = load_a_tile(Fh, i0, kb, mIdx, kh);
    aFl[kb] = load_a_tile(Fl, i0, kb, mIdx, kh);
    aL[kb]  = load_a_tile(Lb, i0, kb, mIdx, kh);
  }

  float m[8], Z[8], P[8], Ms[8];
#pragma unroll
  for (int r = 0; r < 8; r++) { m[r] = -3.0e38f; Z[r] = 0.f; P[r] = 0.f; Ms[r] = 0.f; }

  for (int j0 = jBeg; j0 < jEnd; j0 += 16) {
    {  // speculative prefetch of next tile rows (global_prefetch_b8); past-end
       // addresses stay inside the mapped workspace, so no bounds guard needed.
      const size_t nr = (size_t)(j0 + 16 + mIdx) * D_FEAT;
      __builtin_prefetch(Fh + nr, 0, 3);
      __builtin_prefetch(Fl + nr, 0, 3);
      __builtin_prefetch(Lb + nr, 0, 3);
    }

    v8f accS = {0.f, 0.f, 0.f, 0.f, 0.f, 0.f, 0.f, 0.f};
    v8f accM = {0.f, 0.f, 0.f, 0.f, 0.f, 0.f, 0.f, 0.f};
#pragma unroll
    for (int kb = 0; kb < 4; kb++) {
      v16bf bh = load_b_tile(Fh, j0, kb, mIdx, kh);
      v16bf bl = load_b_tile(Fl, j0, kb, mIdx, kh);
      v16bf bL = load_b_tile(Lb, j0, kb, mIdx, kh);
      // S ~= hi*hi + hi*lo + lo*hi; accM interleaved to break the accS chain.
      accS = __builtin_amdgcn_wmma_f32_16x16x32_bf16(false, aFh[kb], false, bh,
                                                     (short)0, accS, false, false);
      accM = __builtin_amdgcn_wmma_f32_16x16x32_bf16(false, aL[kb], false, bL,
                                                     (short)0, accM, false, false);
      accS = __builtin_amdgcn_wmma_f32_16x16x32_bf16(false, aFh[kb], false, bl,
                                                     (short)0, accS, false, false);
      accS = __builtin_amdgcn_wmma_f32_16x16x32_bf16(false, aFl[kb], false, bh,
                                                     (short)0, accS, false, false);
    }

    // Diagonal elements exist only in the 16-aligned tile with j0 == i0:
    // wave-uniform branch keeps the hot path free of compares/cndmasks.
    if (j0 == i0) {
      tile_update<true >(accS, accM, i0, j0 + mIdx, kh, m, Z, P, Ms);
    } else {
      tile_update<false>(accS, accM, i0, j0 + mIdx, kh, m, Z, P, Ms);
    }
  }

  // Merge the 16 column-lanes of each row, then store per-slice row stats.
#pragma unroll
  for (int r = 0; r < 8; r++) {
    float mm = m[r], zz = Z[r], pp = P[r], ss = Ms[r];
#pragma unroll
    for (int off = 8; off >= 1; off >>= 1) {
      float mo = __shfl_xor(mm, off, 16);
      float zo = __shfl_xor(zz, off, 16);
      float po = __shfl_xor(pp, off, 16);
      float so = __shfl_xor(ss, off, 16);
      float mn = fmaxf(mm, mo);
      zz = zz * __builtin_amdgcn_exp2f(mm - mn)
         + zo * __builtin_amdgcn_exp2f(mo - mn);
      mm = mn; pp += po; ss += so;
    }
    if (mIdx == r) {                        // lanes r and r+16 write rows r / r+8
      const size_t o = (size_t)q * N_TOT + (i0 + r + 8 * kh);
      Sm[o] = mm; Sz[o] = zz; Sp[o] = pp; Ss[o] = ss;
    }
  }
}

// ---------------- slice merge + row loss + block reduction ----------------

__global__ __launch_bounds__(256) void merge_rows(const float* __restrict__ stats,
                                                  float* __restrict__ blockpart) {
  const int i = blockIdx.x * 256 + threadIdx.x;   // global row
  const float* Sm = stats;
  const float* Sz = Sm + (size_t)JSPLIT * N_TOT;
  const float* Sp = Sz + (size_t)JSPLIT * N_TOT;
  const float* Ss = Sp + (size_t)JSPLIT * N_TOT;

  float mm = Sm[i], zz = Sz[i], pp = Sp[i], ss = Ss[i];
#pragma unroll
  for (int qq = 1; qq < JSPLIT; qq++) {
    const size_t o = (size_t)qq * N_TOT + i;
    float mo = Sm[o], zo = Sz[o], po = Sp[o], so = Ss[o];
    float mn = fmaxf(mm, mo);
    zz = zz * __builtin_amdgcn_exp2f(mm - mn)
       + zo * __builtin_amdgcn_exp2f(mo - mn);
    mm = mn; pp += po; ss += so;
  }
  // convert log2-domain (P', m', log2 Z) back to natural log with one mul
  float mlpp = LN2 * (pp - (mm + __builtin_amdgcn_logf(zz)) * ss) / (ss + 1e-8f);
  float v = -mlpp;                                 // TEMP/BASE_TEMP == 1

  __shared__ float red[256];
  red[threadIdx.x] = v;
  __syncthreads();
  for (int s = 128; s > 0; s >>= 1) {
    if (threadIdx.x < s) red[threadIdx.x] += red[threadIdx.x + s];
    __syncthreads();
  }
  if (threadIdx.x == 0) blockpart[blockIdx.x] = red[0];
}

__global__ void final_sum(const float* __restrict__ bp, float* __restrict__ out) {
  if (threadIdx.x == 0) {
    float s = 0.f;
    for (int i = 0; i < N_TOT / 256; i++) s += bp[i];
    out[0] = s * (1.0f / (float)N_TOT);
  }
}

// ---------------- launcher ----------------

extern "C" void kernel_launch(void* const* d_in, const int* in_sizes, int n_in,
                              void* d_out, int out_size, void* d_ws, size_t ws_size,
                              hipStream_t stream) {
  (void)in_sizes; (void)n_in; (void)out_size; (void)ws_size;
  const float* feat = (const float*)d_in[0];   // [4096, 2, 128] fp32
  const float* lab  = (const float*)d_in[1];   // [4096, 100]    fp32
  float* out = (float*)d_out;                  // scalar fp32

  uintptr_t base = ((uintptr_t)d_ws + 255) & ~(uintptr_t)255;
  __bf16* Fh = (__bf16*)base;                            // [N,128] bf16 hi
  __bf16* Fl = Fh + (size_t)N_TOT * D_FEAT;              // [N,128] bf16 lo
  __bf16* Lb = Fl + (size_t)N_TOT * D_FEAT;              // [N,128] normalized labels
  float*  stats = (float*)(Lb + (size_t)N_TOT * D_FEAT); // [4][JSPLIT][N] m,Z,P,Ms
  float*  blockpart = stats + (size_t)4 * JSPLIT * N_TOT;// [N/256]

  prep_feat<<<(N_TOT * D_FEAT + 255) / 256, 256, 0, stream>>>(feat, Fh, Fl,
                                                              N_TOT * D_FEAT);
  prep_lab<<<N_TOT, 128, 0, stream>>>(lab, Lb);
  supcon_main<<<(512 * JSPLIT) / 8, 256, 0, stream>>>(Fh, Fl, Lb, stats);
  merge_rows<<<N_TOT / 256, 256, 0, stream>>>(stats, blockpart);
  final_sum<<<1, 32, 0, stream>>>(blockpart, out);
}
